// GATConv_22093311770838
// MI455X (gfx1250) — matrix operand order repaired and checked by reading, not compiled, not run on hardware
//
#include <hip/hip_runtime.h>
#include <cstdint>

#define NN 4096
#define NF 512
#define NH 8
#define CC 64

typedef __attribute__((ext_vector_type(16))) __bf16 v16bf;
typedef __attribute__((ext_vector_type(8)))  float  v8f;

#if __has_builtin(__builtin_amdgcn_tensor_load_to_lds)
#define HAVE_TDM 1
typedef __attribute__((ext_vector_type(4))) unsigned int u32x4;
typedef __attribute__((ext_vector_type(8))) int          i32x8;
typedef __attribute__((ext_vector_type(4))) int          i32x4;
#endif

__device__ __forceinline__ uint16_t bf16bits(float f) {
    __bf16 h = (__bf16)f;
    uint16_t b;
    __builtin_memcpy(&b, &h, 2);
    return b;
}

// ---------------------------------------------------------------------------
// K1: convert x -> bf16 (row major) and W -> bf16 transposed [out][in]
// ---------------------------------------------------------------------------
__global__ void k_convert(const float* __restrict__ x, const float* __restrict__ W,
                          uint16_t* __restrict__ xb, uint16_t* __restrict__ wT) {
    int idx = blockIdx.x * 256 + threadIdx.x;
    if (idx < NN * NF) xb[idx] = bf16bits(x[idx]);
    if (idx < NF * NF) {
        int k = idx / NF, o = idx % NF;      // W is [in=k][out=o]
        wT[o * NF + k] = bf16bits(W[idx]);
    }
}

// ---------------------------------------------------------------------------
// K2: x_l = x @ W + b   via v_wmma_f32_16x16x32_bf16
// ---------------------------------------------------------------------------
__global__ void __launch_bounds__(256) k_proj(const uint16_t* __restrict__ xb,
                                              const uint16_t* __restrict__ wT,
                                              const float* __restrict__ bias,
                                              float* __restrict__ xl) {
    const int wave = threadIdx.x >> 5, lane = threadIdx.x & 31;
    const int m = lane & 15, hi = lane >> 4;
    const int i0 = blockIdx.x * 16;
    const int n0 = (blockIdx.y * 8 + wave) * 16;

    const uint16_t* arow = xb + (size_t)(i0 + m) * NF;
    const uint16_t* brow = wT + (size_t)(n0 + m) * NF + 16 * hi;

    v8f acc = {};
    for (int k0 = 0; k0 < NF; k0 += 32) {
        union { v16bf v; uint32_t u[8]; } A, B;
#pragma unroll
        for (int vv = 0; vv < 8; ++vv) {
            int ka = 2 * vv + 16 * hi;
            A.u[vv] = *(const uint32_t*)(arow + k0 + ka);
            B.u[vv] = *(const uint32_t*)(brow + k0 + 2 * vv);
        }
        acc = __builtin_amdgcn_wmma_f32_16x16x32_bf16(false, A.v, false, B.v,
                                                      (short)0, acc, false, false);
    }
    float bv = bias[n0 + m];
#pragma unroll
    for (int g = 0; g < 8; ++g) {
        xl[(size_t)(i0 + g + 8 * hi) * NF + n0 + m] = acc[g] + bv;
    }
}

// ---------------------------------------------------------------------------
// K3: alpha_l / alpha_r per (node, head), stored transposed [head][node]
// ---------------------------------------------------------------------------
__global__ void k_alpha(const float* __restrict__ xl, const float* __restrict__ attl,
                        const float* __restrict__ attr,
                        float* __restrict__ alT, float* __restrict__ arT) {
    int wave = threadIdx.x >> 5, lane = threadIdx.x & 31;
    int node = blockIdx.x * 8 + wave;
    const float* row = xl + (size_t)node * NF;
    for (int h = 0; h < NH; ++h) {
        float v0 = row[h * CC + lane], v1 = row[h * CC + lane + 32];
        float sl = v0 * attl[h * CC + lane] + v1 * attl[h * CC + lane + 32];
        float sr = v0 * attr[h * CC + lane] + v1 * attr[h * CC + lane + 32];
#pragma unroll
        for (int off = 16; off >= 1; off >>= 1) {
            sl += __shfl_xor(sl, off, 32);
            sr += __shfl_xor(sr, off, 32);
        }
        if (lane == 0) { alT[h * NN + node] = sl; arT[h * NN + node] = sr; }
    }
}

// ---------------------------------------------------------------------------
// K3b: x_l -> bf16 transposed [feat][node]
// ---------------------------------------------------------------------------
__global__ void k_xlT(const float* __restrict__ xl, uint16_t* __restrict__ xlT) {
    int idx = blockIdx.x * 256 + threadIdx.x;
    int n = idx >> 9, f = idx & (NF - 1);
    xlT[(size_t)f * NN + n] = bf16bits(xl[idx]);
}

// ---------------------------------------------------------------------------
// K4: online-softmax stats per (dst node, head)
// ---------------------------------------------------------------------------
__global__ void k_stats(const int* __restrict__ adj, const float* __restrict__ alT,
                        const float* __restrict__ arT,
                        float* __restrict__ mT, float* __restrict__ zT) {
    int h = threadIdx.x >> 5, lane = threadIdx.x & 31;
    int i = blockIdx.x;
    float al = alT[h * NN + i];
    const int* arow = adj + (size_t)i * NN;
    const float* ar = arT + h * NN;
    float m = -3.0e38f, s = 0.f;
    for (int j = lane; j < NN; j += 32) {
        if (arow[j] > 0) {
            float sc = al * ar[j];
            if (sc > m) { s *= __expf(m - sc); m = sc; }
            s += __expf(sc - m);
        }
    }
#pragma unroll
    for (int off = 16; off >= 1; off >>= 1) {
        float mo = __shfl_xor(m, off, 32);
        float so = __shfl_xor(s, off, 32);
        float M = fmaxf(m, mo);
        s = s * __expf(m - M) + so * __expf(mo - M);
        m = M;
    }
    if (lane == 0) { mT[h * NN + i] = m; zT[h * NN + i] = s; }
}

// ---------------------------------------------------------------------------
// TDM: DMA a 512(row) x 32(col) bf16 tile of xlT into LDS, padded so the LDS
// row pitch is 18 dwords (bank-conflict-free column reads).
//   D# group0: count=1, lds_addr, global_addr, type=2
//   D# group1: data_size=2B, pad every 16 dwords by 2 dwords,
//              tensor 4096 x 512 (stride 4096), tile 32 x 512
// ---------------------------------------------------------------------------
#define ROWPITCH 18                      // dwords per 32-col row in LDS
#ifdef HAVE_TDM
__device__ __forceinline__ void tdm_load_tile(uint32_t lds_off, const uint16_t* gptr) {
    uint64_t ga = (uint64_t)(uintptr_t)gptr;
    u32x4 g0;
    g0[0] = 1u;                                            // count=1, user mode
    g0[1] = lds_off;                                       // lds_addr (bytes)
    g0[2] = (uint32_t)ga;                                  // global_addr[31:0]
    g0[3] = (uint32_t)((ga >> 32) & 0x1FFFFFFu) | 0x80000000u; // addr[56:32] | type=2
    i32x8 g1;
    g1[0] = (int)0x02D10000u;     // data_size=1(2B), pad_en, interval=16dw, amount=2dw
    g1[1] = (int)((NN & 0xFFFF) << 16);                    // tensor_dim0 lo16 (4096)
    g1[2] = (int)((NN >> 16) | ((NF & 0xFFFF) << 16));     // dim0 hi | tensor_dim1 lo (512)
    g1[3] = (int)((NF >> 16) | (32u << 16));               // dim1 hi | tile_dim0 = 32
    g1[4] = (int)NF;                                       // tile_dim1 = 512, tile_dim2 = 0
    g1[5] = (int)NN;                                       // tensor_dim0_stride lo
    g1[6] = 0;
    g1[7] = 0;
    i32x4 z4 = {0, 0, 0, 0};
#if defined(__clang_major__) && (__clang_major__ >= 23)
    i32x8 z8 = {0, 0, 0, 0, 0, 0, 0, 0};
    __builtin_amdgcn_tensor_load_to_lds(g0, g1, z4, z4, z8, 0);
#else
    __builtin_amdgcn_tensor_load_to_lds(g0, g1, z4, z4, 0);
#endif
}
#endif

// ---------------------------------------------------------------------------
// K5: flash-style aggregation. Wave = (16 dst rows, 1 head, all 64 C).
// A (16x32 bf16) = masked softmax numerators computed in registers from adj,
// B (32x16 bf16) = x_l^T tile staged in LDS by the TDM (double-buffered),
// 4 WMMAs per j-tile. Epilogue: /Z, ReLU.
// ---------------------------------------------------------------------------
__global__ void __launch_bounds__(256) k_aggr(const int* __restrict__ adj,
                                              const uint16_t* __restrict__ xlT,
                                              const float* __restrict__ alT,
                                              const float* __restrict__ arT,
                                              const float* __restrict__ mT,
                                              const float* __restrict__ zT,
                                              float* __restrict__ out) {
    const int wave = threadIdx.x >> 5;
    const int h = wave, lane = threadIdx.x & 31;
    const int m = lane & 15, hi = lane >> 4;
    const int i0 = blockIdx.x * 16;

    const float al = alT[h * NN + i0 + m];
    const float mi = mT[h * NN + i0 + m];

    const int*   arow = adj + (size_t)(i0 + m) * NN + 16 * hi;
    const float* arh  = arT + h * NN + 16 * hi;

#ifdef HAVE_TDM
    __shared__ uint32_t buf[2][NF * ROWPITCH];     // 2 x 36 KB padded tiles
    if (wave == 0) tdm_load_tile((uint32_t)(uintptr_t)&buf[0][0], xlT);
#else
    const uint16_t* xt = xlT + (size_t)(h * CC + m) * NN + 16 * hi;
#endif

    v8f acc0 = {}, acc1 = {}, acc2 = {}, acc3 = {};

    for (int jt = 0; jt < NN / 32; ++jt) {
        const int j0 = jt * 32;
#ifdef HAVE_TDM
        if (wave == 0) {
            if (jt + 1 < NN / 32) {
                tdm_load_tile((uint32_t)(uintptr_t)&buf[(jt + 1) & 1][0],
                              xlT + (size_t)(jt + 1) * 32);
                __builtin_amdgcn_s_wait_tensorcnt(1);   // previous tile landed
            } else {
                __builtin_amdgcn_s_wait_tensorcnt(0);
            }
        }
        __syncthreads();                                // tile jt visible to all waves
        const uint32_t* bufc = &buf[jt & 1][0];
#endif

        // A fragment: p = adj ? exp(al*ar - m_i) : 0, packed bf16 pairs
        union { v16bf v; __bf16 e[16]; } A;
#pragma unroll
        for (int vv = 0; vv < 8; ++vv) {
            int kk = 2 * vv;
            int2   aa = *(const int2*)  (arow + j0 + kk);
            float2 rr = *(const float2*)(arh  + j0 + kk);
            float p0 = (aa.x > 0) ? __expf(al * rr.x - mi) : 0.f;
            float p1 = (aa.y > 0) ? __expf(al * rr.y - mi) : 0.f;
            A.e[2 * vv]     = (__bf16)p0;
            A.e[2 * vv + 1] = (__bf16)p1;
        }
        if (j0 + 32 < NN) __builtin_prefetch(arow + j0 + 32, 0, 1);

#pragma unroll
        for (int t = 0; t < 4; ++t) {                   // 4 C-subtiles of 16
            union { v16bf v; uint32_t u[8]; } B;
#ifdef HAVE_TDM
            // feature row r = h*64 + t*16 + col(lane), 8 consecutive dwords/lane
            const uint32_t* p = bufc + (size_t)(h * CC + t * 16 + m) * ROWPITCH + 8 * hi;
#pragma unroll
            for (int vv = 0; vv < 8; ++vv) B.u[vv] = p[vv];
#else
#pragma unroll
            for (int vv = 0; vv < 8; ++vv)
                B.u[vv] = *(const uint32_t*)(xt + (size_t)t * 16 * NN + j0 + 2 * vv);
#endif
            if (t == 0) acc0 = __builtin_amdgcn_wmma_f32_16x16x32_bf16(false, A.v, false, B.v, (short)0, acc0, false, false);
            if (t == 1) acc1 = __builtin_amdgcn_wmma_f32_16x16x32_bf16(false, A.v, false, B.v, (short)0, acc1, false, false);
            if (t == 2) acc2 = __builtin_amdgcn_wmma_f32_16x16x32_bf16(false, A.v, false, B.v, (short)0, acc2, false, false);
            if (t == 3) acc3 = __builtin_amdgcn_wmma_f32_16x16x32_bf16(false, A.v, false, B.v, (short)0, acc3, false, false);
        }
#ifdef HAVE_TDM
        __syncthreads();                                // reads done before overwrite
#endif
    }

    float zr[8];
#pragma unroll
    for (int g = 0; g < 8; ++g) zr[g] = zT[h * NN + i0 + g + 8 * hi];
#pragma unroll
    for (int g = 0; g < 8; ++g) {
        size_t o = (size_t)(i0 + g + 8 * hi) * NF + h * CC + m;
        float inv = 1.0f / zr[g];
        out[o + 0]  = fmaxf(acc0[g] * inv, 0.f);
        out[o + 16] = fmaxf(acc1[g] * inv, 0.f);
        out[o + 32] = fmaxf(acc2[g] * inv, 0.f);
        out[o + 48] = fmaxf(acc3[g] * inv, 0.f);
    }
}

// ---------------------------------------------------------------------------
extern "C" void kernel_launch(void* const* d_in, const int* in_sizes, int n_in,
                              void* d_out, int out_size, void* d_ws, size_t ws_size,
                              hipStream_t stream) {
    const float* x    = (const float*)d_in[0];
    const int*   adj  = (const int*)  d_in[1];
    const float* W    = (const float*)d_in[2];
    const float* b    = (const float*)d_in[3];
    const float* attl = (const float*)d_in[4];
    const float* attr = (const float*)d_in[5];
    float* out = (float*)d_out;

    uint16_t* xb  = (uint16_t*)d_ws;                    // 4096*512 bf16
    uint16_t* wT  = xb + (size_t)NN * NF;               // 512*512 bf16
    float*    xl  = (float*)(wT + (size_t)NF * NF);     // 4096*512 f32
    uint16_t* xlT = (uint16_t*)(xl + (size_t)NN * NF);  // 512*4096 bf16
    float*    alT = (float*)(xlT + (size_t)NF * NN);    // [8][4096]
    float*    arT = alT + NH * NN;
    float*    mT  = arT + NH * NN;
    float*    zT  = mT + NH * NN;

    k_convert<<<dim3((NN * NF) / 256), 256, 0, stream>>>(x, W, xb, wT);
    k_proj   <<<dim3(NN / 16, NF / 128), 256, 0, stream>>>(xb, wT, b, xl);
    k_alpha  <<<dim3(NN / 8), 256, 0, stream>>>(xl, attl, attr, alT, arT);
    k_xlT    <<<dim3((NN * NF) / 256), 256, 0, stream>>>(xl, xlT);
    k_stats  <<<dim3(NN), 256, 0, stream>>>(adj, alT, arT, mT, zT);
    k_aggr   <<<dim3(NN / 16), 256, 0, stream>>>(adj, xlT, alT, arT, mT, zT, out);
}
